// APRE_81166291960476
// MI455X (gfx1250) — matrix operand order repaired and checked by reading, not compiled, not run on hardware
//
#include <hip/hip_runtime.h>
#include <hip/hip_bf16.h>
#include <cmath>

// ---------------------------------------------------------------------------
// APRE-style model, MI455X (gfx1250, wave32, WMMA).
// Heavy GEMM (hidden @ W_out, 34.4 GFLOP/side) done with
// v_wmma_f32_16x16x32_f16, fused with all PAD-axis reductions so the 64MB
// `lang` tensor never touches HBM. Staging vectorized: global_load_b128 ->
// f16 pack -> ds_store_b64.
// ---------------------------------------------------------------------------

typedef __attribute__((ext_vector_type(16))) _Float16 v16h;
typedef __attribute__((ext_vector_type(8)))  _Float16 v8h;
typedef __attribute__((ext_vector_type(4)))  _Float16 v4h;
typedef __attribute__((ext_vector_type(8)))  float    v8f;
typedef __attribute__((ext_vector_type(4)))  float    v4f;

#define TTL_  512   // reviews per side
#define NB_   128   // batch
#define PAD_  128   // padded review length
#define NASP  8     // aspects
#define DD_   256   // feat dim
#define KK_   1024  // bert*nl
#define KC_   32    // K chunk (one WMMA K step)
#define AS_   40    // LDS row stride in halves (16B-aligned, bank-spread)

union Frag16 { v16h v; v8h h[2]; };

__device__ __forceinline__ float wave_red(float v) {
#pragma unroll
  for (int off = 16; off > 0; off >>= 1) v += __shfl_xor(v, off, 32);
  return v;
}

// ---------------------------------------------------------------------------
// Kernel A: one workgroup per review r.
//   lang[p,d] = hidden[r,p,:] @ W_out[:,d] + b_out[d]      (WMMA f16->f32)
//   sum[d]    = sum_p lang ; cnt[d] = #nonzero ; asp[a,d] = sum_p loc*lang
//   cls_repr  = cls[r] @ W_cls ; cat = [cls_repr, avg]
//   sc_ex[a]  = tanh(asp.wex + emb.wex_hi) ; sc_im = tanh(cat.wim)
// ---------------------------------------------------------------------------
__global__ __launch_bounds__(256)
void apre_side_kernel(const float* __restrict__ hidden, const float* __restrict__ cls,
                      const float* __restrict__ loc,    const float* __restrict__ Wout,
                      const float* __restrict__ bout,   const float* __restrict__ Wcls,
                      const float* __restrict__ emb,    const float* __restrict__ wex,
                      const float* __restrict__ wim,
                      float* __restrict__ asp_out, float* __restrict__ cat_out,
                      float* __restrict__ scex_out, float* __restrict__ scim_out)
{
  __shared__ __align__(16) _Float16 Alds[PAD_ * AS_];   // hidden chunk, f16
  __shared__ __align__(16) _Float16 Wlds[DD_ * AS_];    // W_out chunk, transposed f16
  __shared__ float loc_lds[NASP * PAD_];
  __shared__ float bias_lds[DD_];
  __shared__ float cls_lds[DD_];
  __shared__ float sum_lds[DD_];
  __shared__ float cnt_lds[DD_];
  __shared__ float aspacc[NASP * DD_];
  __shared__ float red9[9];

  const int r      = blockIdx.x;
  const int tid    = threadIdx.x;
  const int lane   = tid & 31;
  const int wid    = tid >> 5;          // 8 waves, each owns 16 rows of p
  const int halfid = lane >> 4;
  const int lmod   = lane & 15;

  // ---- init ----
  bias_lds[tid] = bout[tid];
  cls_lds[tid]  = cls[r * DD_ + tid];
  sum_lds[tid]  = 0.f;
  cnt_lds[tid]  = 0.f;
#pragma unroll
  for (int a = 0; a < NASP; ++a) aspacc[a * DD_ + tid] = 0.f;
  for (int i = tid; i < NASP * PAD_; i += 256) loc_lds[i] = loc[(size_t)r * NASP * PAD_ + i];
  if (tid < 9) red9[tid] = 0.f;

  v8f acc[16];
  v8f zero = {};
#pragma unroll
  for (int nt = 0; nt < 16; ++nt) acc[nt] = zero;

  const size_t hbase = (size_t)r * PAD_ * KK_;

  // ---- main K loop: 1024 in chunks of 32 ----
  for (int kc = 0; kc < KK_; kc += KC_) {
    __syncthreads();
    // stage hidden[r, 0..127, kc..kc+31] -> f16 LDS (row-major [p][k])
    // vectorized: float4 load -> pack 4 halves -> ds_store_b64
#pragma unroll
    for (int i = 0; i < 4; ++i) {
      int e  = tid + 256 * i;          // 1024 float4-groups
      int p  = e >> 3;
      int k4 = (e & 7) * 4;
      v4f w  = *(const v4f*)&hidden[hbase + (size_t)p * KK_ + kc + k4];
      v4h h;
      h[0] = (_Float16)w[0]; h[1] = (_Float16)w[1];
      h[2] = (_Float16)w[2]; h[3] = (_Float16)w[3];
      *(v4h*)&Alds[p * AS_ + k4] = h;
    }
    // stage W_out[kc..kc+31, 0..255] -> f16 LDS transposed [n][k]
    // 4x4 register sub-tile transpose: 4 float4 loads -> 4 ds_store_b64
#pragma unroll
    for (int i = 0; i < 2; ++i) {
      int s    = tid + 256 * i;        // 512 sub-tiles of 4(k) x 4(n)
      int st_n = (s & 63) * 4;
      int st_k = (s >> 6) * 4;
      v4f w0 = *(const v4f*)&Wout[(size_t)(kc + st_k + 0) * DD_ + st_n];
      v4f w1 = *(const v4f*)&Wout[(size_t)(kc + st_k + 1) * DD_ + st_n];
      v4f w2 = *(const v4f*)&Wout[(size_t)(kc + st_k + 2) * DD_ + st_n];
      v4f w3 = *(const v4f*)&Wout[(size_t)(kc + st_k + 3) * DD_ + st_n];
#pragma unroll
      for (int j = 0; j < 4; ++j) {
        v4h h;
        h[0] = (_Float16)w0[j]; h[1] = (_Float16)w1[j];
        h[2] = (_Float16)w2[j]; h[3] = (_Float16)w3[j];
        *(v4h*)&Wlds[(st_n + j) * AS_ + st_k] = h;
      }
    }
    // prefetch next chunk of the big hidden stream (global_prefetch_b8)
    if (kc + KC_ < KK_ && tid < PAD_)
      __builtin_prefetch(&hidden[hbase + (size_t)tid * KK_ + kc + KC_], 0, 1);
    __syncthreads();

    // A fragment: 16-bit A 16x32 layout (ISA 7.12.2): lane holds row M=lane&15,
    // VGPR0-3 = K in [kb, kb+8), VGPR4-7 = K in [16+kb, 16+kb+8), kb = (lane>>4)*8
    Frag16 af;
    const int kb   = halfid * 8;
    const int mrow = wid * 16 + lmod;
    af.h[0] = *(const v8h*)&Alds[mrow * AS_ + kb];
    af.h[1] = *(const v8h*)&Alds[mrow * AS_ + 16 + kb];

    // B fragment: lane holds col N=lane&15; VGPR0-3 = K in [kb2, kb2+8), VGPR4-7 = K in [16+kb2, 16+kb2+8), kb2 = (lane>>4)*8
    const int kb2 = halfid * 8;
#pragma unroll
    for (int nt = 0; nt < 16; ++nt) {
      Frag16 bf;
      const int n = nt * 16 + lmod;
      bf.h[0] = *(const v8h*)&Wlds[n * AS_ + kb2];
      bf.h[1] = *(const v8h*)&Wlds[n * AS_ + kb2 + 16];
      acc[nt] = __builtin_amdgcn_wmma_f32_16x16x32_f16(false, af.v, false, bf.v,
                                                       (short)0, acc[nt], false, false);
    }
  }

  // ---- fused PAD-axis reductions on the lang tile ----
  // D layout: lane holds N=lane&15 (+16*nt); VGPR v holds M = v + 8*(lane>>4).
#pragma unroll
  for (int nt = 0; nt < 16; ++nt) {
    const int n     = nt * 16 + lmod;
    const float bia = bias_lds[n];
    float s = 0.f, c = 0.f;
    float aa[NASP];
#pragma unroll
    for (int a = 0; a < NASP; ++a) aa[a] = 0.f;
#pragma unroll
    for (int v = 0; v < 8; ++v) {
      const int p = wid * 16 + halfid * 8 + v;
      float x = acc[nt][v] + bia;
      s += x;
      c += (x != 0.f) ? 1.f : 0.f;
#pragma unroll
      for (int a = 0; a < NASP; ++a) aa[a] += loc_lds[a * PAD_ + p] * x;
    }
    // lanes l and l^16 share the same column n, different p halves
    s += __shfl_xor(s, 16, 32);
    c += __shfl_xor(c, 16, 32);
#pragma unroll
    for (int a = 0; a < NASP; ++a) aa[a] += __shfl_xor(aa[a], 16, 32);
    if (lane < 16) {
      atomicAdd(&sum_lds[n], s);
      atomicAdd(&cnt_lds[n], c);
#pragma unroll
      for (int a = 0; a < NASP; ++a) atomicAdd(&aspacc[a * DD_ + n], aa[a]);
    }
  }
  __syncthreads();

  // ---- per-review tail: thread d = column ----
  const int d = tid;
  float avg  = sum_lds[d] / cnt_lds[d];
  float clsr = 0.f;
  for (int k = 0; k < DD_; ++k) clsr += cls_lds[k] * Wcls[k * DD_ + d];

  cat_out[(size_t)r * 2 * DD_ + d]       = clsr;
  cat_out[(size_t)r * 2 * DD_ + DD_ + d] = avg;
#pragma unroll
  for (int a = 0; a < NASP; ++a)
    asp_out[((size_t)r * NASP + a) * DD_ + d] = aspacc[a * DD_ + d];

  float tim = clsr * wim[d] + avg * wim[DD_ + d];
  tim = wave_red(tim);
  if (lane == 0) atomicAdd(&red9[8], tim);
#pragma unroll
  for (int a = 0; a < NASP; ++a) {
    float t = aspacc[a * DD_ + d] * wex[d] + emb[a * DD_ + d] * wex[DD_ + d];
    t = wave_red(t);
    if (lane == 0) atomicAdd(&red9[a], t);
  }
  __syncthreads();
  if (tid < NASP) scex_out[r * NASP + tid] = tanhf(red9[tid]);   // EX_TEMP == 1
  if (tid == NASP) scim_out[r] = tanhf(red9[8]);
}

// ---------------------------------------------------------------------------
// Kernel B: segment softmax attention (ragged review axis). One WG per example.
// ---------------------------------------------------------------------------
__global__ __launch_bounds__(256)
void apre_segattn_kernel(const float* __restrict__ scex, const float* __restrict__ scim,
                         const float* __restrict__ asp,  const float* __restrict__ cat,
                         const int* __restrict__ seg,
                         float* __restrict__ expl, float* __restrict__ impl)
{
  __shared__ int   sr0, sr1;
  __shared__ float wex_s[NASP][TTL_];
  __shared__ float wim_s[TTL_];

  const int b   = blockIdx.x;
  const int tid = threadIdx.x;
  if (tid == 0) { sr0 = TTL_; sr1 = 0; }
  __syncthreads();
  for (int rr = tid; rr < TTL_; rr += 256)
    if (seg[rr] == b) { atomicMin(&sr0, rr); atomicMax(&sr1, rr + 1); }
  __syncthreads();
  const int r0 = sr0, cnt = sr1 - sr0;   // seg sorted -> contiguous run

  if (tid < NASP) {
    float m = -1e30f;
    for (int j = 0; j < cnt; ++j) m = fmaxf(m, scex[(r0 + j) * NASP + tid]);
    float dsum = 0.f;
    for (int j = 0; j < cnt; ++j) dsum += expf(scex[(r0 + j) * NASP + tid] - m);
    for (int j = 0; j < cnt; ++j)
      wex_s[tid][j] = expf(scex[(r0 + j) * NASP + tid] - m) / dsum;
  } else if (tid == NASP) {
    float m = -1e30f;
    for (int j = 0; j < cnt; ++j) m = fmaxf(m, scim[r0 + j]);
    float dsum = 0.f;
    for (int j = 0; j < cnt; ++j) dsum += expf(scim[r0 + j] - m);
    for (int j = 0; j < cnt; ++j) wim_s[j] = expf(scim[r0 + j] - m) / dsum;
  }
  __syncthreads();

  for (int e = tid; e < NASP * DD_; e += 256) {
    const int a = e >> 8;
    float v = 0.f;
    for (int j = 0; j < cnt; ++j)
      v += wex_s[a][j] * asp[(size_t)(r0 + j) * NASP * DD_ + e];
    expl[(size_t)b * NASP * DD_ + e] = v;
  }
  for (int e = tid; e < 2 * DD_; e += 256) {
    float v = 0.f;
    for (int j = 0; j < cnt; ++j)
      v += wim_s[j] * cat[(size_t)(r0 + j) * 2 * DD_ + e];
    impl[(size_t)b * 2 * DD_ + e] = v;
  }
}

// ---------------------------------------------------------------------------
// Kernel C: final MLPs + bias tables + gamma contraction. One WG per example.
// ---------------------------------------------------------------------------
__global__ __launch_bounds__(256)
void apre_final_kernel(const float* __restrict__ u_expl, const float* __restrict__ i_expl,
                       const float* __restrict__ u_impl, const float* __restrict__ i_impl,
                       const float* __restrict__ exW1, const float* __restrict__ exb1,
                       const float* __restrict__ exW2, const float* __restrict__ imW1,
                       const float* __restrict__ imb1, const float* __restrict__ imW2,
                       const float* __restrict__ bu,   const float* __restrict__ bt,
                       const float* __restrict__ gam,  const int* __restrict__ uid,
                       const int* __restrict__ iid,    float* __restrict__ out)
{
  __shared__ float vec[4 * DD_];
  __shared__ float red;
  __shared__ float result;
  const int b    = blockIdx.x;
  const int d    = threadIdx.x;
  const int lane = d & 31;

  vec[d]       = u_impl[b * 512 + d];
  vec[256 + d] = u_impl[b * 512 + 256 + d];
  vec[512 + d] = i_impl[b * 512 + d];
  vec[768 + d] = i_impl[b * 512 + 256 + d];
  if (d == 0) red = 0.f;
  __syncthreads();

  float h = imb1[d];
  for (int k = 0; k < 1024; ++k) h += vec[k] * imW1[k * DD_ + d];
  h = fmaxf(h, 0.f);
  float t = wave_red(h * imW2[d]);
  if (lane == 0) atomicAdd(&red, t);
  __syncthreads();
  if (d == 0) result = red + bu[uid[b]] + bt[iid[b]];

  for (int a = 0; a < NASP; ++a) {
    __syncthreads();
    vec[d]       = u_expl[(b * NASP + a) * DD_ + d];
    vec[256 + d] = i_expl[(b * NASP + a) * DD_ + d];
    if (d == 0) red = 0.f;
    __syncthreads();
    float h2 = exb1[d];
    for (int k = 0; k < 512; ++k) h2 += vec[k] * exW1[k * DD_ + d];
    h2 = fmaxf(h2, 0.f);
    float t2 = wave_red(h2 * exW2[d]);
    if (lane == 0) atomicAdd(&red, t2);
    __syncthreads();
    if (d == 0) result += red * gam[a];
  }
  __syncthreads();
  if (d == 0) out[b] = result;
}

// ---------------------------------------------------------------------------
extern "C" void kernel_launch(void* const* d_in, const int* in_sizes, int n_in,
                              void* d_out, int out_size, void* d_ws, size_t ws_size,
                              hipStream_t stream)
{
  (void)in_sizes; (void)n_in; (void)out_size; (void)ws_size;

  const float* urevs_hidden = (const float*)d_in[0];
  const float* irevs_hidden = (const float*)d_in[1];
  const float* u_cls        = (const float*)d_in[2];
  const float* i_cls        = (const float*)d_in[3];
  const float* urevs_loc    = (const float*)d_in[4];
  const float* irevs_loc    = (const float*)d_in[5];
  const float* W_u_out      = (const float*)d_in[6];
  const float* b_u_out      = (const float*)d_in[7];
  const float* W_i_out      = (const float*)d_in[8];
  const float* b_i_out      = (const float*)d_in[9];
  const float* emb_aspect   = (const float*)d_in[10];
  const float* w_ex_u       = (const float*)d_in[11];
  const float* w_ex_i       = (const float*)d_in[12];
  const float* w_im_u       = (const float*)d_in[13];
  const float* w_im_i       = (const float*)d_in[14];
  const float* W_ucls       = (const float*)d_in[15];
  const float* W_icls       = (const float*)d_in[16];
  const float* ex_W1        = (const float*)d_in[17];
  const float* ex_b1        = (const float*)d_in[18];
  const float* ex_W2        = (const float*)d_in[19];
  const float* im_W1        = (const float*)d_in[20];
  const float* im_b1        = (const float*)d_in[21];
  const float* im_W2        = (const float*)d_in[22];
  const float* bu_table     = (const float*)d_in[23];
  const float* bt_table     = (const float*)d_in[24];
  const float* gamma        = (const float*)d_in[25];
  const int*   u_seg        = (const int*)d_in[26];
  const int*   i_seg        = (const int*)d_in[27];
  const int*   uid          = (const int*)d_in[28];
  const int*   iid          = (const int*)d_in[29];

  float* ws = (float*)d_ws;
  size_t off = 0;
  float* u_asp  = ws + off; off += (size_t)TTL_ * NASP * DD_;
  float* i_asp  = ws + off; off += (size_t)TTL_ * NASP * DD_;
  float* u_cat  = ws + off; off += (size_t)TTL_ * 2 * DD_;
  float* i_cat  = ws + off; off += (size_t)TTL_ * 2 * DD_;
  float* u_scex = ws + off; off += (size_t)TTL_ * NASP;
  float* i_scex = ws + off; off += (size_t)TTL_ * NASP;
  float* u_scim = ws + off; off += (size_t)TTL_;
  float* i_scim = ws + off; off += (size_t)TTL_;
  float* u_expl = ws + off; off += (size_t)NB_ * NASP * DD_;
  float* i_expl = ws + off; off += (size_t)NB_ * NASP * DD_;
  float* u_impl = ws + off; off += (size_t)NB_ * 2 * DD_;
  float* i_impl = ws + off; off += (size_t)NB_ * 2 * DD_;

  apre_side_kernel<<<TTL_, 256, 0, stream>>>(urevs_hidden, u_cls, urevs_loc, W_u_out,
                                             b_u_out, W_ucls, emb_aspect, w_ex_u, w_im_u,
                                             u_asp, u_cat, u_scex, u_scim);
  apre_side_kernel<<<TTL_, 256, 0, stream>>>(irevs_hidden, i_cls, irevs_loc, W_i_out,
                                             b_i_out, W_icls, emb_aspect, w_ex_i, w_im_i,
                                             i_asp, i_cat, i_scex, i_scim);
  apre_segattn_kernel<<<NB_, 256, 0, stream>>>(u_scex, u_scim, u_asp, u_cat, u_seg,
                                               u_expl, u_impl);
  apre_segattn_kernel<<<NB_, 256, 0, stream>>>(i_scex, i_scim, i_asp, i_cat, i_seg,
                                               i_expl, i_impl);
  apre_final_kernel<<<NB_, 256, 0, stream>>>(u_expl, i_expl, u_impl, i_impl,
                                             ex_W1, ex_b1, ex_W2, im_W1, im_b1, im_W2,
                                             bu_table, bt_table, gamma, uid, iid,
                                             (float*)d_out);
}
